// BaseGGNN_55980603736142
// MI455X (gfx1250) — matrix operand order repaired and checked by reading, not compiled, not run on hardware
//
#include <hip/hip_runtime.h>
#include <hip/hip_bf16.h>

// ---------------------------------------------------------------------------
// GGNN on gfx1250: all GEMMs via v_wmma_f32_16x16x32_bf16 (f32 accumulate).
// Global->LDS staging uses CDNA5 async copies (global_load_async_to_lds_b128,
// ASYNCcnt-tracked) so tile movement runs behind the WMMA stream. All B
// operands stored transposed (N x K) so staging is pure b128 copies.
// Double-buffered LDS, BK=64 (2 WMMA K-substeps per barrier).
// ---------------------------------------------------------------------------

typedef unsigned short u16;
typedef __attribute__((ext_vector_type(16))) __bf16 bf16x16;
typedef __attribute__((ext_vector_type(8)))  __bf16 bf16x8;
typedef __attribute__((ext_vector_type(8)))  float  f32x8;
typedef __attribute__((ext_vector_type(4)))  int    v4i;
typedef __attribute__((address_space(1))) v4i* gv4i_p;   // global int4*
typedef __attribute__((address_space(3))) v4i* lv4i_p;   // LDS int4*

#define NNODES 2048
#define DDIM   512
#define E2     8                    // 2 * N_EDGE_TYPES
#define KADJ   (NNODES * E2)        // 16384
#define ANNDIM 256
#define NSTEPS 5

__device__ __forceinline__ u16 f2bf(float f) {
  union { float f; unsigned u; } cv; cv.f = f;
  unsigned u = cv.u;
  return (u16)((u + 0x7FFFu + ((u >> 16) & 1u)) >> 16);   // RNE
}
__device__ __forceinline__ float sigmoid_f(float x) { return 1.0f / (1.0f + expf(-x)); }

// ---- CDNA5 async global->LDS copy (ASYNCcnt) ------------------------------
__device__ __forceinline__ void async_cp_b128(const u16* g, u16* l) {
#if __has_builtin(__builtin_amdgcn_global_load_async_to_lds_b128)
  __builtin_amdgcn_global_load_async_to_lds_b128((gv4i_p)(g), (lv4i_p)(l), 0, 0);
#else
  unsigned loff = (unsigned)(unsigned long long)(const void*)l;   // LDS offset
  unsigned long long ga = (unsigned long long)(const void*)g;
  asm volatile("global_load_async_to_lds_b128 %0, %1, off"
               :: "v"(loff), "v"(ga) : "memory");
#endif
}
__device__ __forceinline__ void wait_async0() {
#if __has_builtin(__builtin_amdgcn_s_wait_asynccnt)
  __builtin_amdgcn_s_wait_asynccnt(0);
#else
  asm volatile("s_wait_asynccnt 0" ::: "memory");
#endif
}

// ------------------------------- utility kernels ---------------------------

__global__ void k_cvt_bf16(const float* __restrict__ s, u16* __restrict__ d, size_t n) {
  size_t i  = (size_t)blockIdx.x * blockDim.x + threadIdx.x;
  size_t st = (size_t)gridDim.x * blockDim.x;
  for (; i < n; i += st) d[i] = f2bf(s[i]);
}

// src: nBatch matrices [K][N] row-major  ->  dst: nBatch matrices [N][K] (B^T)
__global__ void k_cvt_bf16_T(const float* __restrict__ s, u16* __restrict__ d,
                             int K, int N, size_t total) {
  size_t i  = (size_t)blockIdx.x * blockDim.x + threadIdx.x;
  size_t st = (size_t)gridDim.x * blockDim.x;
  const size_t mat = (size_t)K * N;
  for (; i < total; i += st) {
    size_t b = i / mat, rem = i - b * mat;
    int k = (int)(rem / N), n = (int)(rem - (size_t)k * N);
    d[b * mat + (size_t)n * K + k] = f2bf(s[i]);
  }
}

__global__ void k_init_h(const float* __restrict__ ann, float* __restrict__ h32,
                         u16* __restrict__ hbf) {
  int idx = blockIdx.x * blockDim.x + threadIdx.x;      // over NNODES*DDIM
  int row = idx >> 9, d = idx & (DDIM - 1);
  float v = (d < ANNDIM) ? ann[row * ANNDIM + d] : 0.0f;
  h32[idx] = v;
  hbf[idx] = f2bf(v);
}

__global__ void k_copy_f32(const float* __restrict__ s, float* __restrict__ d, size_t n) {
  size_t i = (size_t)blockIdx.x * blockDim.x + threadIdx.x;
  if (i < n) d[i] = s[i];
}

// ------------------------------- WMMA GEMM ---------------------------------
// C[M,N] = sum_pairs A_p[M,K] * B_p[K,N].  A row-major [M][K]; B supplied
// TRANSPOSED as [N][K] (leading dim K). Block tile 64(M) x 128(N), BK=64;
// 128 threads = 4 wave32 in 2x2 grid, each wave a 32x64 tile = 2x4 fragments
// -> 16 wmma per wave per staged tile (2 K-substeps of 32).
// Async global->LDS double buffering: one s_wait_asynccnt + barrier per tile.
//
// Epilogue modes:
//  0: transposed bf16 out (packed b128): outBF[gn*ldOutT + bz*M + gm]
//     = bf16(acc + bias[gn])                                  (T^T = (hW+b)^T)
//  1: outF = acc ; outBF = bf16(acc)                          (a = Adj*T)
//  2: outF = sigmoid(acc + bias[n])                           (z)
//  3: outBF = bf16( sigmoid(acc + bias[n]) * aux0[idx] )      (r*h)
//  4: outF = acc + bias[n]                                    (ah = aWh+bh)
//  5: ht = tanh(acc + aux0); h = (1-aux1)*outF + aux1*ht ;
//     outF = h ; outBF = bf16(h)                              (GRU update)
template <int EPI, int NPAIRS>
__global__ __launch_bounds__(128) void wmma_gemm(
    const u16* __restrict__ A0, const u16* __restrict__ B0,
    const u16* __restrict__ A1, const u16* __restrict__ B1,
    int M, int N, int K, int ldOutT,
    const float* __restrict__ bias,
    float* __restrict__ outF, u16* __restrict__ outBF,
    const float* __restrict__ aux0, const float* __restrict__ aux1) {
  constexpr int BM = 64, BN = 128, BK = 64, PAD = 8;
  __shared__ __align__(16) u16 sA[2][BM][BK + PAD];   // [m][k]
  __shared__ __align__(16) u16 sB[2][BN][BK + PAD];   // [n][k] (B^T rows)

  const int tid    = threadIdx.x;
  const int lane   = tid & 31;
  const int w      = tid >> 5;
  const int laneLo = lane & 15;
  const int laneHi = lane >> 4;
  const int wm     = (w & 1) * 32;    // wave tile: 32 rows
  const int wn     = (w >> 1) * 64;   //            64 cols
  const int tileM  = blockIdx.x * BM;
  const int tileN  = blockIdx.y * BN;
  const size_t bz  = blockIdx.z;

  const u16*   Bz    = B0 + bz * (size_t)K * N;
  const float* biasz = bias ? (bias + bz * (size_t)N) : nullptr;

  // 64x64 A tile: 4096 elems -> 4 async b128 per thread
  auto stageA = [&](const u16* Ap, int kb, int buf) {
#pragma unroll
    for (int p2 = 0; p2 < 4; ++p2) {
      const int q  = tid + p2 * 128;      // 0..511
      const int r  = q >> 3;              // row 0..63
      const int kc = (q & 7) * 8;         // 0..56
      async_cp_b128(Ap + (size_t)(tileM + r) * K + kb + kc, &sA[buf][r][kc]);
    }
  };
  // 128x64 B^T tile: 8192 elems -> 8 async b128 per thread
  auto stageB = [&](const u16* Bp, int kb, int buf) {
#pragma unroll
    for (int p2 = 0; p2 < 8; ++p2) {
      const int q  = tid + p2 * 128;      // 0..1023
      const int r  = q >> 3;              // B^T row (n) 0..127
      const int kc = (q & 7) * 8;
      async_cp_b128(Bp + (size_t)(tileN + r) * K + kb + kc, &sB[buf][r][kc]);
    }
  };

  f32x8 acc[2][4] = {};

  for (int pair = 0; pair < NPAIRS; ++pair) {
    const u16* Ap = pair ? A1 : A0;
    const u16* Bp = pair ? B1 : Bz;

    __syncthreads();                       // buffer reuse safety across pairs
    stageA(Ap, 0, 0);
    stageB(Bp, 0, 0);

    int p = 0;
    for (int kb = 0; kb < K; kb += BK) {
      wait_async0();                       // this wave's copies into lds[p] done
      __syncthreads();                     // all waves' copies done; reads of p^1 done
      if (kb + BK < K) {                   // stage next tile behind the WMMAs
        stageA(Ap, kb + BK, p ^ 1);
        stageB(Bp, kb + BK, p ^ 1);
      }

#pragma unroll
      for (int sub = 0; sub < 2; ++sub) {  // two 16x16x32 K-substeps
        const int ko = sub * 32;
        bf16x16 afr[2], bfr[4];
#pragma unroll
        for (int fm = 0; fm < 2; ++fm) {
          const int arow = wm + fm * 16 + laneLo;
          const bf16x8 lo = *reinterpret_cast<const bf16x8*>(&sA[p][arow][ko + laneHi * 8]);
          const bf16x8 hi = *reinterpret_cast<const bf16x8*>(&sA[p][arow][ko + laneHi * 8 + 16]);
#pragma unroll
          for (int i = 0; i < 8; ++i) { afr[fm][i] = lo[i]; afr[fm][i + 8] = hi[i]; }
        }
#pragma unroll
        for (int fn = 0; fn < 4; ++fn) {
          const int brow = wn + fn * 16 + laneLo;
          const bf16x8 lo = *reinterpret_cast<const bf16x8*>(&sB[p][brow][ko + laneHi * 8]);
          const bf16x8 hi = *reinterpret_cast<const bf16x8*>(&sB[p][brow][ko + laneHi * 8 + 16]);
#pragma unroll
          for (int i = 0; i < 8; ++i) { bfr[fn][i] = lo[i]; bfr[fn][i + 8] = hi[i]; }
        }
#pragma unroll
        for (int fm = 0; fm < 2; ++fm)
#pragma unroll
          for (int fn = 0; fn < 4; ++fn)
            acc[fm][fn] = __builtin_amdgcn_wmma_f32_16x16x32_bf16(
                false, afr[fm], false, bfr[fn], (short)0, acc[fm][fn], false, false);
      }
      p ^= 1;
    }
  }

  // ---- epilogue: C/D layout row = vgpr + laneHi*8, col = laneLo ----
#pragma unroll
  for (int fm = 0; fm < 2; ++fm)
#pragma unroll
    for (int fn = 0; fn < 4; ++fn) {
      const int gn     = tileN + wn + fn * 16 + laneLo;
      const int gmBase = tileM + wm + fm * 16 + laneHi * 8;
      if (EPI == 0) {
        // transposed packed store: 8 consecutive rows -> one b128
        union { uint4 v; u16 s[8]; } pk;
        const float b = biasz[gn];
#pragma unroll
        for (int j = 0; j < 8; ++j) pk.s[j] = f2bf(acc[fm][fn][j] + b);
        *reinterpret_cast<uint4*>(
            outBF + (size_t)gn * ldOutT + bz * (size_t)M + gmBase) = pk.v;
      } else {
#pragma unroll
        for (int j = 0; j < 8; ++j) {
          const size_t idx = (size_t)(gmBase + j) * N + gn;
          float v = acc[fm][fn][j];
          if (EPI == 1) {
            outF[idx]  = v;
            outBF[idx] = f2bf(v);
          } else if (EPI == 2) {
            outF[idx] = sigmoid_f(v + biasz[gn]);
          } else if (EPI == 3) {
            outBF[idx] = f2bf(sigmoid_f(v + biasz[gn]) * aux0[idx]);
          } else if (EPI == 4) {
            outF[idx] = v + biasz[gn];
          } else {  // EPI == 5: GRU update, in-place h
            const float ht   = tanhf(v + aux0[idx]);
            const float z    = aux1[idx];
            const float hnew = (1.0f - z) * outF[idx] + z * ht;
            outF[idx]  = hnew;
            outBF[idx] = f2bf(hnew);
          }
        }
      }
    }
}

// ------------------------------- launch ------------------------------------

extern "C" void kernel_launch(void* const* d_in, const int* in_sizes, int n_in,
                              void* d_out, int out_size, void* d_ws, size_t ws_size,
                              hipStream_t stream) {
  const float* adjacency   = (const float*)d_in[0];   // [2048, 16384]
  const float* annotations = (const float*)d_in[1];   // [2048, 256]
  const float* W_prop      = (const float*)d_in[2];   // [8, 512, 512]
  const float* b_prop      = (const float*)d_in[3];   // [8, 512]
  const float* Wz = (const float*)d_in[4];
  const float* Uz = (const float*)d_in[5];
  const float* bz = (const float*)d_in[6];
  const float* Wr = (const float*)d_in[7];
  const float* Ur = (const float*)d_in[8];
  const float* br = (const float*)d_in[9];
  const float* Wh = (const float*)d_in[10];
  const float* Uh = (const float*)d_in[11];
  const float* bh = (const float*)d_in[12];
  float* out = (float*)d_out;

  // ---- workspace carve-out (~115 MB total) ----
  char*  ws  = (char*)d_ws;
  size_t off = 0;
  auto carve = [&](size_t bytes) -> void* {
    void* p = ws + off;
    off += (bytes + 255) & ~(size_t)255;
    return p;
  };
  const size_t ND = (size_t)NNODES * DDIM;            // 1,048,576
  const size_t WW = (size_t)DDIM * DDIM;
  u16*   adjbf = (u16*)  carve((size_t)NNODES * KADJ * 2);   // A op, row-major
  u16*   wpbfT = (u16*)  carve((size_t)E2 * WW * 2);         // B ops, transposed
  u16*   wzbfT = (u16*)  carve(WW * 2);
  u16*   uzbfT = (u16*)  carve(WW * 2);
  u16*   wrbfT = (u16*)  carve(WW * 2);
  u16*   urbfT = (u16*)  carve(WW * 2);
  u16*   whbfT = (u16*)  carve(WW * 2);
  u16*   uhbfT = (u16*)  carve(WW * 2);
  float* h32   = (float*)carve(ND * 4);
  u16*   hbf   = (u16*)  carve(ND * 2);
  u16*   TbfT  = (u16*)  carve((size_t)E2 * ND * 2);   // T^T: [512][16384]
  float* a32   = (float*)carve(ND * 4);
  u16*   abf   = (u16*)  carve(ND * 2);
  float* z32   = (float*)carve(ND * 4);
  u16*   rhbf  = (u16*)  carve(ND * 2);
  float* ah32  = (float*)carve(ND * 4);
  (void)ws_size; (void)in_sizes; (void)n_in; (void)out_size;

  // ---- one-time conversions (weights -> transposed bf16) ----
  k_cvt_bf16<<<2048, 256, 0, stream>>>(adjacency, adjbf, (size_t)NNODES * KADJ);
  k_cvt_bf16_T<<<1024, 256, 0, stream>>>(W_prop, wpbfT, DDIM, DDIM, (size_t)E2 * WW);
  k_cvt_bf16_T<<<512, 256, 0, stream>>>(Wz, wzbfT, DDIM, DDIM, WW);
  k_cvt_bf16_T<<<512, 256, 0, stream>>>(Uz, uzbfT, DDIM, DDIM, WW);
  k_cvt_bf16_T<<<512, 256, 0, stream>>>(Wr, wrbfT, DDIM, DDIM, WW);
  k_cvt_bf16_T<<<512, 256, 0, stream>>>(Ur, urbfT, DDIM, DDIM, WW);
  k_cvt_bf16_T<<<512, 256, 0, stream>>>(Wh, whbfT, DDIM, DDIM, WW);
  k_cvt_bf16_T<<<512, 256, 0, stream>>>(Uh, uhbfT, DDIM, DDIM, WW);
  k_init_h<<<(int)(ND / 256), 256, 0, stream>>>(annotations, h32, hbf);

  const dim3 blk(128);
  const dim3 gNB(NNODES / 64, DDIM / 128, 1);     // 32 x 4
  const dim3 gT (NNODES / 64, DDIM / 128, E2);    // batched over edge types

  for (int step = 0; step < NSTEPS; ++step) {
    // T^T[f, e*2048+n'] = (h @ W_prop[e] + b_prop[e])^T   (bf16, packed b128)
    wmma_gemm<0, 1><<<gT, blk, 0, stream>>>(
        hbf, wpbfT, nullptr, nullptr, NNODES, DDIM, DDIM, KADJ,
        b_prop, nullptr, TbfT, nullptr, nullptr);
    // a = Adj @ T    (B given as T^T [512][16384])
    wmma_gemm<1, 1><<<gNB, blk, 0, stream>>>(
        adjbf, TbfT, nullptr, nullptr, NNODES, DDIM, KADJ, 0,
        nullptr, a32, abf, nullptr, nullptr);
    // z = sigmoid(a@Wz + h@Uz + bz)
    wmma_gemm<2, 2><<<gNB, blk, 0, stream>>>(
        abf, wzbfT, hbf, uzbfT, NNODES, DDIM, DDIM, 0,
        bz, z32, nullptr, nullptr, nullptr);
    // rh = sigmoid(a@Wr + h@Ur + br) * h    (bf16)
    wmma_gemm<3, 2><<<gNB, blk, 0, stream>>>(
        abf, wrbfT, hbf, urbfT, NNODES, DDIM, DDIM, 0,
        br, nullptr, rhbf, h32, nullptr);
    // ah = a@Wh + bh
    wmma_gemm<4, 1><<<gNB, blk, 0, stream>>>(
        abf, whbfT, nullptr, nullptr, NNODES, DDIM, DDIM, 0,
        bh, ah32, nullptr, nullptr, nullptr);
    // h = (1-z)*h + z*tanh(rh@Uh + ah)   (updates h32 + hbf in place)
    wmma_gemm<5, 1><<<gNB, blk, 0, stream>>>(
        rhbf, uhbfT, nullptr, nullptr, NNODES, DDIM, DDIM, 0,
        nullptr, h32, hbf, ah32, z32);
  }

  k_copy_f32<<<(int)(ND / 256), 256, 0, stream>>>(h32, out, ND);
}
